// CMDB_51367808860420
// MI455X (gfx1250) — matrix operand order repaired and checked by reading, not compiled, not run on hardware
//
#include <hip/hip_runtime.h>
#include <hip/hip_bf16.h>
#include <math.h>

#define BN   4
#define DIMC 64
#define HH   88
#define WW   88
#define LL   (HH*WW)          // 7744
#define C2   (2*DIMC)         // 128
#define NST  64
#define DH   32               // dim/2
#define RR   2                // max(DH/16,1)
#define PROJR (RR + 2*NST)    // 130
#define HP   90               // padded H
#define WP   90               // padded W
#define LP   (HP*WP)          // 8100

typedef __attribute__((ext_vector_type(16))) _Float16 v16h;
typedef __attribute__((ext_vector_type(8)))  float    v8f;

__device__ __forceinline__ float sigmoidf_(float x){ return 1.f/(1.f+__expf(-x)); }
__device__ __forceinline__ float siluf_(float x){ return x*sigmoidf_(x); }

// ---------------- WMMA GEMM: Y[b,o,l] = sum_c W[o,c]*X[b,c,l] (+bias)(+resid) ----------------
// block = 64 threads (2 waves); each wave computes a 16(M) x 32(L) tile (two accumulators
// sharing one A tile). K fully unrolled -> all global loads use immediate offsets.
template<int K>
__global__ void gemm_wmma_kernel(const float* __restrict__ Wm, const float* __restrict__ X,
                                 const float* __restrict__ bias, const float* __restrict__ resid,
                                 float* __restrict__ Y,
                                 int O, int L,
                                 long x_bstride, int x_row0,
                                 long y_bstride, int y_row0,
                                 long r_bstride)
{
  const int lane = threadIdx.x & 31;
  const int wave = threadIdx.x >> 5;
  const int l0   = (blockIdx.x * (blockDim.x >> 5) + wave) * 32;
  const int m0   = blockIdx.y * 16;
  const int b    = blockIdx.z;
  const int row  = lane & 15;
  const int g    = lane >> 4;
  const int m    = m0 + row;
  const int mc   = (m < O) ? m : (O - 1);       // clamp: invalid rows never stored
  const float* Wr  = Wm + (long)mc * K;
  const float* Xb0 = X + (long)b * x_bstride + (long)x_row0 * L + (l0 + row);
  const float* Xb1 = Xb0 + 16;

  v8f acc0 = {}, acc1 = {};
#pragma unroll
  for (int k0 = 0; k0 < K; k0 += 32) {
    v16h av, bv0, bv1;
#pragma unroll
    for (int e = 0; e < 16; ++e) {
      const int v = e >> 1, p = e & 1;
      const int ka = ((v & 4) << 2) + g * 8 + ((v & 3) << 1) + p;  // A 16x32 f16 layout
      const int kb = ((e & 8) << 1) + g * 8 + (e & 7);              // B 32x16 f16 layout
      av[e]  = (_Float16)Wr[k0 + ka];
      bv0[e] = (_Float16)Xb0[(long)(k0 + kb) * L];
      bv1[e] = (_Float16)Xb1[(long)(k0 + kb) * L];
    }
    acc0 = __builtin_amdgcn_wmma_f32_16x16x32_f16(false, av, false, bv0, (short)0, acc0, false, false);
    acc1 = __builtin_amdgcn_wmma_f32_16x16x32_f16(false, av, false, bv1, (short)0, acc1, false, false);
  }
  const int ncol = lane & 15;
  const int mg   = g * 8;
#pragma unroll
  for (int r = 0; r < 8; ++r) {
    const int mm = m0 + r + mg;
    if (mm < O) {
      float v0 = acc0[r], v1 = acc1[r];
      if (bias)  { const float bb = bias[mm]; v0 += bb; v1 += bb; }
      if (resid) {
        const float* rp = resid + (long)b * r_bstride + (long)mm * L + (l0 + ncol);
        v0 += rp[0]; v1 += rp[16];
      }
      float* yp = Y + (long)b * y_bstride + (long)(y_row0 + mm) * L + (l0 + ncol);
      yp[0] = v0; yp[16] = v1;
    }
  }
}

// ---------------- zero-pad (B,128,88,88) -> (B,128,90,90), border = 0 ----------------
__global__ void pad_kernel(const float* __restrict__ X, float* __restrict__ Y)
{
  const long n = (long)BN * C2 * LP;
  for (long i = (long)blockIdx.x * blockDim.x + threadIdx.x; i < n; i += (long)gridDim.x * blockDim.x) {
    const int wp = (int)(i % WP); long t = i / WP;
    const int hp = (int)(t % HP); t /= HP;
    const int c  = (int)(t % C2); const int b = (int)(t / C2);
    const int h = hp - 1, w = wp - 1;
    const bool inb = (unsigned)h < (unsigned)HH && (unsigned)w < (unsigned)WW;
    const int hc = inb ? h : 0, wc = inb ? w : 0;
    const float v = X[((long)b * C2 + c) * LL + hc * WW + wc];
    Y[i] = inb ? v : 0.f;
  }
}

// ---------------- 3x3 conv (pad=1) on pre-padded input ----------------
// Runtime tap loop (9 iters, no unroll -> bounded VGPR pressure, no spills);
// fully-unrolled K=128 channel-GEMM per tap (4 k-steps x 2 WMMA).
__global__ void conv3x3_wmma_kernel(const float* __restrict__ Wc, const float* __restrict__ Xp,
                                    float* __restrict__ Y)
{
  const int L = LL;
  const int lane = threadIdx.x & 31;
  const int wave = threadIdx.x >> 5;
  const int l0 = (blockIdx.x * (blockDim.x >> 5) + wave) * 32;
  const int m0 = blockIdx.y * 16;
  const int b  = blockIdx.z;
  const int row = lane & 15;
  const int g   = lane >> 4;
  const int Kt  = C2 * 9;
  const int m   = m0 + row;
  const int pl0 = l0 + row,        ph0 = pl0 / WW, pw0 = pl0 % WW;
  const int pl1 = pl0 + 16,        ph1 = pl1 / WW, pw1 = pl1 % WW;
  const float* Wr  = Wc + (long)m * Kt;
  const float* Xpb = Xp + (long)b * C2 * LP;

  v8f acc0 = {}, acc1 = {};
#pragma unroll 1
  for (int tap = 0; tap < 9; ++tap) {
    const int dh = tap / 3, dw = tap - dh * 3;       // 0..2 over padded coords
    const float* Wt  = Wr + tap;
    const float* Xt0 = Xpb + (long)(ph0 + dh) * WP + (pw0 + dw);
    const float* Xt1 = Xpb + (long)(ph1 + dh) * WP + (pw1 + dw);
#pragma unroll
    for (int k0 = 0; k0 < C2; k0 += 32) {
      v16h av, bv0, bv1;
#pragma unroll
      for (int e = 0; e < 16; ++e) {
        const int v = e >> 1, p = e & 1;
        const int ka = ((v & 4) << 2) + g * 8 + ((v & 3) << 1) + p;
        const int kb = ((e & 8) << 1) + g * 8 + (e & 7);
        av[e]  = (_Float16)Wt[(k0 + ka) * 9];
        bv0[e] = (_Float16)Xt0[(long)(k0 + kb) * LP];
        bv1[e] = (_Float16)Xt1[(long)(k0 + kb) * LP];
      }
      acc0 = __builtin_amdgcn_wmma_f32_16x16x32_f16(false, av, false, bv0, (short)0, acc0, false, false);
      acc1 = __builtin_amdgcn_wmma_f32_16x16x32_f16(false, av, false, bv1, (short)0, acc1, false, false);
    }
  }
  const int ncol = lane & 15, mg = g * 8;
#pragma unroll
  for (int r = 0; r < 8; ++r) {
    float* yp = Y + ((long)b * C2 + (m0 + r + mg)) * L + (l0 + ncol);
    yp[0] = acc0[r]; yp[16] = acc1[r];
  }
}

// ---------------- BatchNorm ----------------
__global__ void bn_stats_kernel(const float* __restrict__ X, float* __restrict__ mean,
                                float* __restrict__ rstd, int C, int L)
{
  const int c = blockIdx.x;
  __shared__ float s1[256], s2[256];
  float sum = 0.f, sq = 0.f;
  const long n = (long)BN * L;
  for (long i = threadIdx.x; i < n; i += blockDim.x) {
    const int b = (int)(i / L); const int l = (int)(i - (long)b * L);
    const float v = X[((long)b * C + c) * L + l];
    sum += v; sq += v * v;
  }
  s1[threadIdx.x] = sum; s2[threadIdx.x] = sq;
  __syncthreads();
  for (int s = blockDim.x >> 1; s > 0; s >>= 1) {
    if ((int)threadIdx.x < s) { s1[threadIdx.x] += s1[threadIdx.x+s]; s2[threadIdx.x] += s2[threadIdx.x+s]; }
    __syncthreads();
  }
  if (threadIdx.x == 0) {
    const float mu = s1[0] / (float)n;
    const float var = s2[0] / (float)n - mu * mu;
    mean[c] = mu; rstd[c] = rsqrtf(var + 1e-5f);
  }
}

__global__ void bn_apply_kernel(const float* __restrict__ X, const float* __restrict__ mean,
                                const float* __restrict__ rstd, const float* __restrict__ gw,
                                const float* __restrict__ bw, float* __restrict__ Y,
                                int C, int L, int relu)
{
  const long n = (long)BN * C * L;
  for (long i = (long)blockIdx.x * blockDim.x + threadIdx.x; i < n; i += (long)gridDim.x * blockDim.x) {
    const int c = (int)((i / L) % C);
    float v = gw[c] * (X[i] - mean[c]) * rstd[c] + bw[c];
    if (relu) v = fmaxf(v, 0.f);
    Y[i] = v;
  }
}

// ---------------- depthwise causal conv1d (K=4) + SiLU ----------------
__global__ void dwconv_silu_kernel(const float* __restrict__ X, const float* __restrict__ w,
                                   const float* __restrict__ cb, float* __restrict__ Y, int L)
{
  const long n = (long)BN * DH * L;
  for (long i = (long)blockIdx.x * blockDim.x + threadIdx.x; i < n; i += (long)gridDim.x * blockDim.x) {
    const int l = (int)(i % L); const long t = i / L; const int d = (int)(t % DH); const int b = (int)(t / DH);
    const float* xp = X + ((long)b * DH + d) * L;
    float s = cb[d];
#pragma unroll
    for (int j = 0; j < 4; ++j) {
      const int li = l - 3 + j;
      const float xv = xp[li < 0 ? 0 : li];           // unconditional load
      s += (li >= 0 ? w[d*4+j] : 0.f) * xv;
    }
    Y[i] = siluf_(s);
  }
}

// ---------------- dt = softplus(dtr @ dt_w + dt_b), R=2 ----------------
__global__ void dt_softplus_kernel(const float* __restrict__ proj, const float* __restrict__ dtw,
                                   const float* __restrict__ dtb0, float* __restrict__ Y, int L)
{
  const long n = (long)BN * DH * L;
  for (long i = (long)blockIdx.x * blockDim.x + threadIdx.x; i < n; i += (long)gridDim.x * blockDim.x) {
    const int l = (int)(i % L); const long t = i / L; const int d = (int)(t % DH); const int b = (int)(t / DH);
    const float p0 = proj[((long)b * PROJR + 0) * L + l];
    const float p1 = proj[((long)b * PROJR + 1) * L + l];
    const float x = dtw[d*2+0]*p0 + dtw[d*2+1]*p1 + dtb0[d];
    Y[i] = (x > 20.f) ? x : log1pf(__expf(x));
  }
}

// ---------------- SSM scan: one wave per (b,d); lane holds states n=lane, lane+32 ----------------
__global__ void ssm_scan_kernel(const float* __restrict__ xc, const float* __restrict__ dtv,
                                const float* __restrict__ proj, const float* __restrict__ Alog,
                                const float* __restrict__ Dsk, const float* __restrict__ yz,
                                float* __restrict__ outp, int L)
{
  const int bd = blockIdx.x; const int b = bd / DH, d = bd % DH;
  const int lane = threadIdx.x;
  const float A0 = -__expf(Alog[d*NST + lane]);
  const float A1 = -__expf(Alog[d*NST + lane + 32]);
  const float* Bm0 = proj + ((long)b*PROJR + RR + lane) * L;
  const float* Bm1 = Bm0 + 32L * L;
  const float* Cm0 = proj + ((long)b*PROJR + RR + NST + lane) * L;
  const float* Cm1 = Cm0 + 32L * L;
  const float* xp = xc  + ((long)b*DH + d) * L;
  const float* dp = dtv + ((long)b*DH + d) * L;
  const float* zp = yz  + ((long)b*DIMC + d) * L;   // z = y[:, :DH]
  float* op = outp + ((long)b*DH + d) * L;
  const float dsk = Dsk[d];
  float h0 = 0.f, h1 = 0.f;
  for (int l = 0; l < L; ++l) {
    const float dt = dp[l], xv = xp[l];
    const float dbu = dt * xv;
    h0 = __expf(dt * A0) * h0 + dbu * Bm0[l];
    h1 = __expf(dt * A1) * h1 + dbu * Bm1[l];
    float v = h0 * Cm0[l] + h1 * Cm1[l];
    for (int off = 16; off > 0; off >>= 1) v += __shfl_xor(v, off, 32);
    if (lane == 0) {
      const float z = zp[l];
      op[l] = (v + dsk * xv) * siluf_(z);
    }
  }
}

// ---------------- reverse along L ----------------
__global__ void reverse_rows_kernel(const float* __restrict__ src, float* __restrict__ dst,
                                    int C, int L, long sb, long db)
{
  const long n = (long)BN * C * L;
  for (long i = (long)blockIdx.x * blockDim.x + threadIdx.x; i < n; i += (long)gridDim.x * blockDim.x) {
    const int l = (int)(i % L); const long t = i / L; const int c = (int)(t % C); const int b = (int)(t / C);
    dst[(long)b*db + (long)c*L + l] = src[(long)b*sb + (long)c*L + (L-1-l)];
  }
}

// ---------------- CBAM ----------------
__global__ void chan_pool_kernel(const float* __restrict__ X, long bstride,
                                 float* __restrict__ avg, float* __restrict__ mx, int C, int L)
{
  const int bc = blockIdx.x; const int b = bc / C, c = bc % C;
  const float* xp = X + (long)b * bstride + (long)c * L;
  __shared__ float s1[256], s2[256];
  float sum = 0.f, m = -3.4e38f;
  for (int l = threadIdx.x; l < L; l += blockDim.x) { const float v = xp[l]; sum += v; m = fmaxf(m, v); }
  s1[threadIdx.x] = sum; s2[threadIdx.x] = m;
  __syncthreads();
  for (int s = blockDim.x >> 1; s > 0; s >>= 1) {
    if ((int)threadIdx.x < s) { s1[threadIdx.x] += s1[threadIdx.x+s]; s2[threadIdx.x] = fmaxf(s2[threadIdx.x], s2[threadIdx.x+s]); }
    __syncthreads();
  }
  if (threadIdx.x == 0) { avg[bc] = s1[0] / (float)L; mx[bc] = s2[0]; }
}

__global__ void cbam_mlp_kernel(const float* __restrict__ avg, const float* __restrict__ mx,
                                const float* __restrict__ w1, const float* __restrict__ w2,
                                float* __restrict__ ca)
{
  const int b = blockIdx.x, c = threadIdx.x;
  __shared__ float ha[4], hm[4];
  if (c < 4) {
    float sa_ = 0.f, sm_ = 0.f;
    for (int j = 0; j < DIMC; ++j) { sa_ += w1[c*DIMC+j]*avg[b*DIMC+j]; sm_ += w1[c*DIMC+j]*mx[b*DIMC+j]; }
    ha[c] = fmaxf(sa_, 0.f); hm[c] = fmaxf(sm_, 0.f);
  }
  __syncthreads();
  float s = 0.f;
  for (int r = 0; r < 4; ++r) s += w2[c*4+r] * (ha[r] + hm[r]);
  ca[b*DIMC+c] = sigmoidf_(s);
}

__global__ void spatial_pool_kernel(const float* __restrict__ X, long bstride,
                                    const float* __restrict__ ca, float* __restrict__ sp, int L)
{
  const long n = (long)BN * L;
  for (long i = (long)blockIdx.x * blockDim.x + threadIdx.x; i < n; i += (long)gridDim.x * blockDim.x) {
    const int b = (int)(i / L); const int l = (int)(i % L);
    float sum = 0.f, m = -3.4e38f;
    for (int c = 0; c < DIMC; ++c) {
      const float v = X[(long)b*bstride + (long)c*L + l] * ca[b*DIMC+c];
      sum += v; m = fmaxf(m, v);
    }
    sp[((long)b*2 + 0)*L + l] = sum / (float)DIMC;
    sp[((long)b*2 + 1)*L + l] = m;
  }
}

__global__ void conv7_sigmoid_kernel(const float* __restrict__ sp, const float* __restrict__ spw,
                                     float* __restrict__ sa)
{
  const long n = (long)BN * LL;
  for (long i = (long)blockIdx.x * blockDim.x + threadIdx.x; i < n; i += (long)gridDim.x * blockDim.x) {
    const int b = (int)(i / LL); const int l = (int)(i % LL);
    const int h = l / WW, w = l % WW;
    float s = 0.f;
    for (int ch = 0; ch < 2; ++ch)
      for (int u = 0; u < 7; ++u)
        for (int v = 0; v < 7; ++v) {
          const int hs = h + u - 3, ws = w + v - 3;
          const bool inb = (unsigned)hs < (unsigned)HH && (unsigned)ws < (unsigned)WW;
          const int hc = inb ? hs : 0, wc = inb ? ws : 0;
          const float xv = sp[((long)b*2 + ch)*LL + hc*WW + wc];     // unconditional load
          s += inb ? spw[(ch*7+u)*7+v] * xv : 0.f;
        }
    sa[i] = sigmoidf_(s);
  }
}

__global__ void cbam_apply_kernel(float* __restrict__ X, long bstride,
                                  const float* __restrict__ ca, const float* __restrict__ sa, int L)
{
  const long n = (long)BN * DIMC * L;
  for (long i = (long)blockIdx.x * blockDim.x + threadIdx.x; i < n; i += (long)gridDim.x * blockDim.x) {
    const int l = (int)(i % L); const long t = i / L; const int c = (int)(t % DIMC); const int b = (int)(t / DIMC);
    X[(long)b*bstride + (long)c*L + l] *= ca[b*DIMC+c] * sa[(long)b*L + l];
  }
}

// ================= host =================
extern "C" void kernel_launch(void* const* d_in, const int* in_sizes, int n_in,
                              void* d_out, int out_size, void* d_ws, size_t ws_size,
                              hipStream_t stream)
{
  (void)in_sizes; (void)n_in; (void)out_size; (void)ws_size;
  auto in = [&](int i){ return (const float*)d_in[i]; };
  float* ws = (float*)d_ws;
  const int L = LL;
  const long SZ32 = (long)BN*DH*L, SZ64 = (long)BN*DIMC*L, SZ128 = (long)BN*C2*L, SZ130 = (long)BN*PROJR*L;

  float* nr   = ws;
  float* nd   = nr   + SZ64;
  float* rgbd = nd   + SZ64;
  float* cmx  = rgbd + SZ128;
  float* cmy  = cmx  + SZ32;
  float* cmxr = cmy  + SZ64;
  float* cmyr = cmxr + SZ32;
  float* proj = cmyr + SZ64;
  float* xc   = proj + SZ130;
  float* dtb  = xc   + SZ32;
  float* ssm  = dtb  + SZ32;
  float* tmp  = ssm  + SZ32;
  float* conc = tmp  + SZ32;
  float* smal = conc + SZ64;
  float* sp   = smal + 1024;
  float* sa   = sp   + (long)BN*2*L;
  float* n3   = cmx;      // overlay: cmx..cmyr region is dead by FFN time (needs SZ128 <= 2*(SZ32+SZ64))
  float* xpad = xc;       // overlay: xc..conc region is dead by FFN time (needs BN*C2*LP <= 4*SZ32+SZ64)
  float* hid  = proj;     // overlay (SZ128 <= SZ130)

  float* mean = smal, *rstd = smal+128, *avgp = smal+256, *mxp = smal+512, *ca = smal+768;

  const dim3 blk256(256);
  auto gs = [&](long n){ long g = (n + 255) / 256; if (g > 65535) g = 65535; return dim3((unsigned)g); };

  auto gemm = [&](const float* Wm, const float* X, const float* bias, const float* resid,
                  float* Y, int O, int K, long xbs, int xr0, long ybs, int yr0, long rbs){
    dim3 grid(L/64, (O+15)/16, BN); dim3 blk(64);
    if (K == 32)
      gemm_wmma_kernel<32><<<grid, blk, 0, stream>>>(Wm, X, bias, resid, Y, O, L, xbs, xr0, ybs, yr0, rbs);
    else if (K == 64)
      gemm_wmma_kernel<64><<<grid, blk, 0, stream>>>(Wm, X, bias, resid, Y, O, L, xbs, xr0, ybs, yr0, rbs);
    else
      gemm_wmma_kernel<128><<<grid, blk, 0, stream>>>(Wm, X, bias, resid, Y, O, L, xbs, xr0, ybs, yr0, rbs);
  };

  auto bn = [&](const float* X, const float* g, const float* b, float* Y, int C, int relu){
    bn_stats_kernel<<<dim3(C), blk256, 0, stream>>>(X, mean, rstd, C, L);
    bn_apply_kernel<<<gs((long)BN*C*L), blk256, 0, stream>>>(X, mean, rstd, g, b, Y, C, L, relu);
  };

  // param base pb: conv_w, conv_b, xy_w, dt_w, dt_b, A_log, Dsk, out_w
  auto cms6 = [&](int pb, const float* x, const float* y, float* dst, long dbs, int dr0){
    dwconv_silu_kernel<<<gs(SZ32), blk256, 0, stream>>>(x, in(pb+0), in(pb+1), xc, L);
    gemm(in(pb+2), y, nullptr, nullptr, proj, PROJR, DH, (long)DIMC*L, DH, (long)PROJR*L, 0, 0);
    dt_softplus_kernel<<<gs(SZ32), blk256, 0, stream>>>(proj, in(pb+3), in(pb+4), dtb, L);
    ssm_scan_kernel<<<dim3(BN*DH), dim3(32), 0, stream>>>(xc, dtb, proj, in(pb+5), in(pb+6), y, ssm, L);
    gemm(in(pb+7), ssm, nullptr, nullptr, dst, DH, DH, (long)DH*L, 0, dbs, dr0, 0);
  };

  // pb: to_x_w, to_x_b, to_y_w, to_y_b, mF[8], mR[8], proj_w, proj_b
  auto cmfb = [&](int pb, const float* xin, const float* yin, const float* resid, int r0){
    gemm(in(pb+0), xin, in(pb+1), nullptr, cmx, DH,   DIMC, (long)DIMC*L, 0, (long)DH*L,   0, 0);
    gemm(in(pb+2), yin, in(pb+3), nullptr, cmy, DIMC, DIMC, (long)DIMC*L, 0, (long)DIMC*L, 0, 0);
    cms6(pb+4, cmx, cmy, conc, (long)DIMC*L, 0);                       // xF -> rows 0..31
    reverse_rows_kernel<<<gs(SZ32), blk256, 0, stream>>>(cmx, cmxr, DH,   L, (long)DH*L,   (long)DH*L);
    reverse_rows_kernel<<<gs(SZ64), blk256, 0, stream>>>(cmy, cmyr, DIMC, L, (long)DIMC*L, (long)DIMC*L);
    cms6(pb+12, cmxr, cmyr, tmp, (long)DH*L, 0);
    reverse_rows_kernel<<<gs(SZ32), blk256, 0, stream>>>(tmp, conc + (long)DH*L, DH, L, (long)DH*L, (long)DIMC*L); // xR -> rows 32..63
    gemm(in(pb+20), conc, in(pb+21), resid, rgbd, DIMC, DIMC, (long)DIMC*L, 0, (long)C2*L, r0, (long)DIMC*L);
  };

  auto cbam = [&](int pb, int r0){
    float* xbase = rgbd + (long)r0 * L;  // row offset within (B,128,L)
    chan_pool_kernel<<<dim3(BN*DIMC), blk256, 0, stream>>>(xbase, (long)C2*L, avgp, mxp, DIMC, L);
    cbam_mlp_kernel<<<dim3(BN), dim3(64), 0, stream>>>(avgp, mxp, in(pb+0), in(pb+1), ca);
    spatial_pool_kernel<<<gs((long)BN*L), blk256, 0, stream>>>(xbase, (long)C2*L, ca, sp, L);
    conv7_sigmoid_kernel<<<gs((long)BN*L), blk256, 0, stream>>>(sp, in(pb+2), sa);
    cbam_apply_kernel<<<gs(SZ64), blk256, 0, stream>>>(xbase, (long)C2*L, ca, sa, L);
  };

  const float* rgb = in(0); const float* dep = in(1);
  bn(rgb, in(2), in(3), nr, DIMC, 0);
  bn(dep, in(4), in(5), nd, DIMC, 0);
  cmfb(6,  nr, nd, rgb, 0);      // d2r -> rgbd rows 0..63   (residual = rgb)
  cmfb(28, nd, nr, dep, DIMC);   // r2d -> rgbd rows 64..127 (residual = depth)
  cbam(50, 0);
  cbam(53, DIMC);
  // FFN: rgbd + conv1x1(relu(bn(conv3x3(bn(rgbd)))))
  bn(rgbd, in(56), in(57), n3, C2, 0);
  pad_kernel<<<gs((long)BN*C2*LP), blk256, 0, stream>>>(n3, xpad);
  conv3x3_wmma_kernel<<<dim3(L/64, C2/16, BN), dim3(64), 0, stream>>>(in(58), xpad, hid);
  bn(hid, in(59), in(60), hid, C2, 1);
  gemm(in(61), hid, in(62), rgbd, (float*)d_out, C2, C2, (long)C2*L, 0, (long)C2*L, 0, (long)C2*L);
}
